// DeepSeekMLA_11785390260702
// MI455X (gfx1250) — matrix operand order repaired and checked by reading, not compiled, not run on hardware
//
#include <hip/hip_runtime.h>
#include <hip/hip_bf16.h>
#include <math.h>

typedef __bf16 bf16;
typedef __attribute__((ext_vector_type(16))) __bf16 v16bf;
typedef __attribute__((ext_vector_type(8)))  __bf16 v8bf;
typedef __attribute__((ext_vector_type(8)))  float  v8f;

#define DEV __device__ __forceinline__

// ---------------------------------------------------------------------------
// Cross-half (lane ^ 16) swap: v_permlanex16_b32 with identity selects is a
// pure-VALU xor-16 on wave32 (no LDS round-trip like ds_bpermute).
// ---------------------------------------------------------------------------
DEV int xor16i(int x) {
#if __has_builtin(__builtin_amdgcn_permlanex16)
  return __builtin_amdgcn_permlanex16(x, x, 0x76543210, 0xfedcba98, false, false);
#else
  return __shfl_xor(x, 16, 32);
#endif
}
DEV float xor16f(float x) { return __int_as_float(xor16i(__float_as_int(x))); }

// ---------------------------------------------------------------------------
// WMMA helpers (CDNA5 16x16x32 bf16, f32 accumulate)
// ---------------------------------------------------------------------------
DEV v8f wmma_bf16(v16bf a, v16bf b, v8f c) {
  return __builtin_amdgcn_wmma_f32_16x16x32_bf16(false, a, false, b, (short)0, c,
                                                 false, false);
}

// A-fragment 16x32 (MxK): lane&15 = row M; lanes 0-15 hold K[k0..k0+15],
// lanes 16-31 hold K[k0+16..k0+31] (contiguous along K).
DEV v16bf frag_a(const bf16* A, int lda, int m0, int k0, int lane) {
  const bf16* p = A + (size_t)(m0 + (lane & 15)) * lda + k0 + ((lane >> 4) << 4);
  v8bf lo = *(const v8bf*)p;
  v8bf hb = *(const v8bf*)(p + 8);
  v16bf f;
#pragma unroll
  for (int i = 0; i < 8; ++i) { f[i] = lo[i]; f[i + 8] = hb[i]; }
  return f;
}
// Same fragment but converting from f32 source on the fly (float4 loads).
DEV v16bf frag_a(const float* A, int lda, int m0, int k0, int lane) {
  const float* p = A + (size_t)(m0 + (lane & 15)) * lda + k0 + ((lane >> 4) << 4);
  float4 a0 = *(const float4*)(p);
  float4 a1 = *(const float4*)(p + 4);
  float4 b0 = *(const float4*)(p + 8);
  float4 b1 = *(const float4*)(p + 12);
  v16bf f;
  f[0] = (bf16)a0.x; f[1] = (bf16)a0.y; f[2] = (bf16)a0.z; f[3] = (bf16)a0.w;
  f[4] = (bf16)a1.x; f[5] = (bf16)a1.y; f[6] = (bf16)a1.z; f[7] = (bf16)a1.w;
  f[8] = (bf16)b0.x; f[9] = (bf16)b0.y; f[10] = (bf16)b0.z; f[11] = (bf16)b0.w;
  f[12] = (bf16)b1.x; f[13] = (bf16)b1.y; f[14] = (bf16)b1.z; f[15] = (bf16)b1.w;
  return f;
}
// B-fragment 32x16 (KxN) from transposed storage BT (N x K row-major):
// lane&15 = col N; lanes 0-15 hold K[k0..k0+15], lanes 16-31 K[k0+16..k0+31],
// contiguous along K -> single 32B load.
DEV v16bf frag_b(const bf16* BT, int ldb, int n0, int k0, int lane) {
  const bf16* p = BT + (size_t)(n0 + (lane & 15)) * ldb + k0 + ((lane >> 4) << 4);
  return *(const v16bf*)p;
}

// ---------------------------------------------------------------------------
// Generic WMMA GEMM: C[M,N] = A[M,K] * BT[N,K]^T.  One 32x32 tile per wave
// (2 A-frags x 2 B-frags -> 4 WMMAs per k-step, 1 frag load per WMMA),
// 4 waves / 128-thread block.  Output mapping modes:
//   0: row-major C[row*N+col]
//   1: per-head scatter  C[((b*H+h)*S+s)*64 + offs + d]   (h=col/hd, d=col%hd)
//   2: transposed V      C[((b*H+h)*64+d)*S + s]          (h=col/64, d=col%64)
// ---------------------------------------------------------------------------
template <typename AT, typename OT>
__global__ void wmma_gemm_kernel(const AT* __restrict__ A, const bf16* __restrict__ BT,
                                 OT* __restrict__ C, int M, int N, int K,
                                 int mode, int hd, int offs, int Sdim, int Hdim) {
  int lane = threadIdx.x & 31;
  int wid = blockIdx.x * (blockDim.x >> 5) + (threadIdx.x >> 5);
  int tn = N >> 5;
  int m0 = (wid / tn) << 5;
  int n0 = (wid % tn) << 5;
  v8f acc00 = {}, acc01 = {}, acc10 = {}, acc11 = {};
  for (int k0 = 0; k0 < K; k0 += 32) {
    if (k0 + 32 < K) {  // global_prefetch_b8 of next A chunks
      __builtin_prefetch(A + (size_t)(m0 + (lane & 15)) * K + k0 + 32, 0, 3);
      __builtin_prefetch(A + (size_t)(m0 + 16 + (lane & 15)) * K + k0 + 32, 0, 3);
    }
    v16bf a0 = frag_a(A, K, m0, k0, lane);
    v16bf a1 = frag_a(A, K, m0 + 16, k0, lane);
    v16bf b0 = frag_b(BT, K, n0, k0, lane);
    v16bf b1 = frag_b(BT, K, n0 + 16, k0, lane);
    acc00 = wmma_bf16(a0, b0, acc00);
    acc01 = wmma_bf16(a0, b1, acc01);
    acc10 = wmma_bf16(a1, b0, acc10);
    acc11 = wmma_bf16(a1, b1, acc11);
  }
  int hi = lane >> 4, cl = lane & 15;
  v8f accs[2][2] = {{acc00, acc01}, {acc10, acc11}};
#pragma unroll
  for (int tm = 0; tm < 2; ++tm) {
#pragma unroll
    for (int tc = 0; tc < 2; ++tc) {
#pragma unroll
      for (int r = 0; r < 8; ++r) {
        int row = m0 + (tm << 4) + r + (hi << 3);
        int col = n0 + (tc << 4) + cl;
        float v = accs[tm][tc][r];
        if (mode == 0) {
          C[(size_t)row * N + col] = (OT)v;
        } else if (mode == 1) {
          int bb = row / Sdim, ss = row % Sdim;
          int h = col / hd, d = col % hd;
          C[(((size_t)(bb * Hdim + h) * Sdim + ss) << 6) + offs + d] = (OT)v;
        } else {
          int bb = row / Sdim, ss = row % Sdim;
          int h = col >> 6, d = col & 63;
          C[(((size_t)(bb * Hdim + h) << 6) + d) * Sdim + ss] = (OT)v;
        }
      }
    }
  }
}

// f32 (K x N) -> bf16 transposed (N x K)
__global__ void transpose_to_bf16_kernel(const float* __restrict__ in,
                                         bf16* __restrict__ out, int K, int N) {
  int t = blockIdx.x * blockDim.x + threadIdx.x;
  if (t >= K * N) return;
  int n = t / K, k = t % K;
  out[t] = (bf16)in[(size_t)k * N + n];
}

// RMS-norm rows of length 128: one wave per row, 4 elems per lane.
__global__ void rmsnorm_kernel(const float* __restrict__ X, const float* __restrict__ W,
                               bf16* __restrict__ Y, int rows) {
  int wid = (blockIdx.x * blockDim.x + threadIdx.x) >> 5;
  int lane = threadIdx.x & 31;
  if (wid >= rows) return;
  const float4 v = *(const float4*)(X + (size_t)wid * 128 + lane * 4);
  float ss = v.x * v.x + v.y * v.y + v.z * v.z + v.w * v.w;
#pragma unroll
  for (int m = 1; m < 16; m <<= 1) ss += __shfl_xor(ss, m, 32);
  ss += xor16f(ss);
  float rs = rsqrtf(ss * (1.0f / 128.0f) + 1e-5f);
  const float4 w = *(const float4*)(W + lane * 4);
  bf16* out = Y + (size_t)wid * 128 + lane * 4;
  out[0] = (bf16)(v.x * rs * w.x);
  out[1] = (bf16)(v.y * rs * w.y);
  out[2] = (bf16)(v.z * rs * w.z);
  out[3] = (bf16)(v.w * rs * w.w);
}

// RoPE in-place on the rope half (dims 32..63) of Q and K, layout (B,H,S,64).
// One thread per (b,h,s,d<16): handles the rotation pair (d, d+16).
__global__ void rope_kernel(bf16* __restrict__ Q, bf16* __restrict__ K,
                            const float* __restrict__ cosb,
                            const float* __restrict__ sinb, int Sdim) {
  int t = blockIdx.x * blockDim.x + threadIdx.x;
  int d = t & 15;
  int site = t >> 4;  // b*H*S + h*S + s
  int s = site % Sdim;
  size_t base = (size_t)site * 64 + 32;
  float c1 = cosb[s * 32 + d], s1 = sinb[s * 32 + d];
  float c2 = cosb[s * 32 + 16 + d], s2 = sinb[s * 32 + 16 + d];
  float q1 = (float)Q[base + d], q2 = (float)Q[base + 16 + d];
  Q[base + d] = (bf16)(q1 * c1 - q2 * s1);
  Q[base + 16 + d] = (bf16)(q2 * c2 + q1 * s2);
  float k1 = (float)K[base + d], k2 = (float)K[base + 16 + d];
  K[base + d] = (bf16)(k1 * c1 - k2 * s1);
  K[base + 16 + d] = (bf16)(k2 * c2 + k1 * s2);
}

// ---------------------------------------------------------------------------
// Flash attention, causal.  Q,K: (B,H,S,64) bf16; VT: (B,H,64,S) bf16.
// Transposed formulation: ST = K * Q^T puts q across lanes and keys in the
// VGPR index, so the softmax row-reduction is in-lane + ONE permlanex16,
// m/l/alpha are per-lane scalars, and O^T = V^T * P^T needs no LDS at all:
// the P^T B-fragment is built from registers + 4 permlanex16 swaps.
// The key loop is split into an unmasked main loop and a masked diagonal
// tail (MASKED template bool) so v_cmp/v_cndmask only run on 1-2 blocks.
// ---------------------------------------------------------------------------
template <bool MASKED>
DEV void attn_block(const bf16* __restrict__ Kh, const bf16* __restrict__ Vh,
                    v16bf qb0, v16bf qb1, int k0, int qabs, int hi, int lane,
                    float& m, float& l, v8f& o0, v8f& o1, v8f& o2, v8f& o3,
                    int Sdim) {
  const float scale = 0.125f;  // 1/sqrt(64)
  // ST[key, q] for keys k0..k0+31 (two 16-key C tiles)
  v8f st0 = {}, st1 = {};
  st0 = wmma_bf16(frag_a(Kh, 64, k0, 0, lane), qb0, st0);
  st0 = wmma_bf16(frag_a(Kh, 64, k0, 32, lane), qb1, st0);
  st1 = wmma_bf16(frag_a(Kh, 64, k0 + 16, 0, lane), qb0, st1);
  st1 = wmma_bf16(frag_a(Kh, 64, k0 + 16, 32, lane), qb1, st1);

  float a0[8], a1[8], bm = -INFINITY;
#pragma unroll
  for (int r = 0; r < 8; ++r) {
    if (MASKED) {
      int key0 = k0 + r + (hi << 3);  // key index of st0[r] on this lane
      a0[r] = (key0 > qabs) ? -1e9f : st0[r] * scale;
      a1[r] = (key0 + 16 > qabs) ? -1e9f : st1[r] * scale;
    } else {
      a0[r] = st0[r] * scale;
      a1[r] = st1[r] * scale;
    }
    bm = fmaxf(bm, fmaxf(a0[r], a1[r]));
  }
  bm = fmaxf(bm, xor16f(bm));  // combine the two key-halves
  float mn = fmaxf(m, bm);
  float al = __expf(m - mn);
  m = mn;

  union { v8bf v; int i[4]; } X, Y, Hop, P;
  float ps = 0.f;
#pragma unroll
  for (int r = 0; r < 8; ++r) {
    float p0 = __expf(a0[r] - mn);
    float p1 = __expf(a1[r] - mn);
    ps += p0 + p1;
    X.v[r] = (bf16)p0;  // keys k0 + r + 8*hi
    Y.v[r] = (bf16)p1;  // keys k0 + 16 + r + 8*hi
  }
  ps += xor16f(ps);
  l = l * al + ps;

  // Build P^T B-fragment (32 keys x 16 q): each half only needs the partner's
  // same-tag vector, so swap the opposite-tag selection once (4 permlanes).
#pragma unroll
  for (int j = 0; j < 4; ++j) Hop.i[j] = hi ? X.i[j] : Y.i[j];
#pragma unroll
  for (int j = 0; j < 4; ++j) P.i[j] = xor16i(Hop.i[j]);  // hi=0: Xp, hi=1: Yp
  union { v16bf v; int i[8]; } pb;
#pragma unroll
  for (int j = 0; j < 4; ++j) {
    pb.i[j]     = hi ? P.i[j] : X.i[j];  // keys hi*16 + 0..7
    pb.i[4 + j] = hi ? Y.i[j] : P.i[j];  // keys hi*16 + 8..15
  }

#pragma unroll
  for (int r = 0; r < 8; ++r) {
    o0[r] *= al; o1[r] *= al; o2[r] *= al; o3[r] *= al;
  }
  o0 = wmma_bf16(frag_a(Vh, Sdim, 0, k0, lane),  pb.v, o0);
  o1 = wmma_bf16(frag_a(Vh, Sdim, 16, k0, lane), pb.v, o1);
  o2 = wmma_bf16(frag_a(Vh, Sdim, 32, k0, lane), pb.v, o2);
  o3 = wmma_bf16(frag_a(Vh, Sdim, 48, k0, lane), pb.v, o3);
}

__global__ void mla_attention_kernel(const bf16* __restrict__ Q, const bf16* __restrict__ K,
                                     const bf16* __restrict__ VT, bf16* __restrict__ O,
                                     int Sdim, int Hdim) {
  int lane = threadIdx.x & 31;
  int w = threadIdx.x >> 5;
  int hi = lane >> 4, cl = lane & 15;
  int q0 = (blockIdx.x * (blockDim.x >> 5) + w) << 4;
  int bh = blockIdx.y;
  const bf16* Qh = Q + (size_t)bh * Sdim * 64;
  const bf16* Kh = K + (size_t)bh * Sdim * 64;
  const bf16* Vh = VT + (size_t)bh * 64 * Sdim;

  // Q as the B matrix (d x q), loaded once per wave.
  v16bf qb0 = frag_b(Qh, 64, q0, 0, lane);
  v16bf qb1 = frag_b(Qh, 64, q0, 32, lane);

  v8f o0 = {}, o1 = {}, o2 = {}, o3 = {};  // O^T tiles: d = 16t + r + 8*hi, q = cl
  float m = -INFINITY, l = 0.f;            // per-lane state for q = q0 + cl
  int qabs = q0 + cl;

  // Blocks entirely below the diagonal (key k0+31 <= q0): no masking needed.
  int kfull = (q0 >= 31) ? ((q0 - 31) & ~31) + 32 : 0;  // first masked k0
  int k0 = 0;
  for (; k0 < kfull; k0 += 32)
    attn_block<false>(Kh, Vh, qb0, qb1, k0, qabs, hi, lane, m, l, o0, o1, o2, o3, Sdim);
  for (; k0 <= q0 + 15; k0 += 32)
    attn_block<true>(Kh, Vh, qb0, qb1, k0, qabs, hi, lane, m, l, o0, o1, o2, o3, Sdim);

  float inv = 1.0f / l;
  int bb = bh / Hdim, h = bh % Hdim;
  // lane writes 8 contiguous d's per tile: d = 16t + 8*hi + r
  size_t base = (((size_t)(bb * Sdim + q0 + cl) * Hdim + h) << 6) + (hi << 3);
  v8bf ov;
#pragma unroll
  for (int r = 0; r < 8; ++r) ov[r] = (bf16)(o0[r] * inv);
  *(v8bf*)(O + base) = ov;
#pragma unroll
  for (int r = 0; r < 8; ++r) ov[r] = (bf16)(o1[r] * inv);
  *(v8bf*)(O + base + 16) = ov;
#pragma unroll
  for (int r = 0; r < 8; ++r) ov[r] = (bf16)(o2[r] * inv);
  *(v8bf*)(O + base + 32) = ov;
#pragma unroll
  for (int r = 0; r < 8; ++r) ov[r] = (bf16)(o3[r] * inv);
  *(v8bf*)(O + base + 48) = ov;
}

// ---------------------------------------------------------------------------
// Host side
// ---------------------------------------------------------------------------
extern "C" void kernel_launch(void* const* d_in, const int* in_sizes, int n_in,
                              void* d_out, int out_size, void* d_ws, size_t ws_size,
                              hipStream_t stream) {
  (void)in_sizes; (void)n_in; (void)out_size; (void)ws_size;
  constexpr int Bb = 2, Ss = 2048, Dd = 2048, Hh = 16, Rr = 128;
  constexpr int Mrows = Bb * Ss;  // 4096

  const float* x         = (const float*)d_in[0];
  const float* cosb      = (const float*)d_in[1];
  const float* sinb      = (const float*)d_in[2];
  /* d_in[3] mask: causal tril(0,-1e9) applied analytically in-kernel */
  const float* w_kv_down = (const float*)d_in[4];
  const float* kv_norm_w = (const float*)d_in[5];
  const float* w_uk      = (const float*)d_in[6];
  const float* w_ur      = (const float*)d_in[7];
  const float* w_uv      = (const float*)d_in[8];
  const float* w_q_down  = (const float*)d_in[9];
  const float* q_norm_w  = (const float*)d_in[10];
  const float* w_uq      = (const float*)d_in[11];
  const float* w_qr      = (const float*)d_in[12];
  const float* w_o       = (const float*)d_in[13];
  float* out = (float*)d_out;

  char* ws = (char*)d_ws;
  size_t off = 0;
  auto alloc = [&](size_t bytes) {
    void* p = ws + off;
    off = (off + bytes + 255) & ~(size_t)255;
    return p;
  };
  bf16*  wkvdT  = (bf16*)alloc((size_t)Rr * Dd * 2);
  bf16*  wqdT   = (bf16*)alloc((size_t)Rr * Dd * 2);
  bf16*  wukT   = (bf16*)alloc((size_t)512 * Rr * 2);
  bf16*  wurT   = (bf16*)alloc((size_t)512 * Rr * 2);
  bf16*  wuvT   = (bf16*)alloc((size_t)1024 * Rr * 2);
  bf16*  wuqT   = (bf16*)alloc((size_t)512 * Rr * 2);
  bf16*  wqrT   = (bf16*)alloc((size_t)512 * Rr * 2);
  bf16*  woT    = (bf16*)alloc((size_t)Dd * 1024 * 2);
  float* ckvraw = (float*)alloc((size_t)Mrows * Rr * 4);
  float* cqraw  = (float*)alloc((size_t)Mrows * Rr * 4);
  bf16*  ckv    = (bf16*)alloc((size_t)Mrows * Rr * 2);
  bf16*  cq     = (bf16*)alloc((size_t)Mrows * Rr * 2);
  bf16*  Qb     = (bf16*)alloc((size_t)Bb * Hh * Ss * 64 * 2);
  bf16*  Kb     = (bf16*)alloc((size_t)Bb * Hh * Ss * 64 * 2);
  bf16*  VTb    = (bf16*)alloc((size_t)Bb * Hh * 64 * Ss * 2);
  bf16*  attout = (bf16*)alloc((size_t)Mrows * 1024 * 2);

  auto tgrid = [](int n) { return (n + 255) / 256; };
  // Weight transposition + bf16 conversion (K x N -> N x K)
  transpose_to_bf16_kernel<<<tgrid(Dd * Rr), 256, 0, stream>>>(w_kv_down, wkvdT, Dd, Rr);
  transpose_to_bf16_kernel<<<tgrid(Dd * Rr), 256, 0, stream>>>(w_q_down, wqdT, Dd, Rr);
  transpose_to_bf16_kernel<<<tgrid(Rr * 512), 256, 0, stream>>>(w_uk, wukT, Rr, 512);
  transpose_to_bf16_kernel<<<tgrid(Rr * 512), 256, 0, stream>>>(w_ur, wurT, Rr, 512);
  transpose_to_bf16_kernel<<<tgrid(Rr * 1024), 256, 0, stream>>>(w_uv, wuvT, Rr, 1024);
  transpose_to_bf16_kernel<<<tgrid(Rr * 512), 256, 0, stream>>>(w_uq, wuqT, Rr, 512);
  transpose_to_bf16_kernel<<<tgrid(Rr * 512), 256, 0, stream>>>(w_qr, wqrT, Rr, 512);
  transpose_to_bf16_kernel<<<tgrid(1024 * Dd), 256, 0, stream>>>(w_o, woT, 1024, Dd);

  // 32x32 tile per wave, 4 waves per block
  auto gblocks = [](int M_, int N_) { return (M_ / 32) * (N_ / 32) / 4; };
  // Down projections (A = f32 x, converted on the fly)
  wmma_gemm_kernel<float, float><<<gblocks(Mrows, Rr), 128, 0, stream>>>(
      x, wkvdT, ckvraw, Mrows, Rr, Dd, 0, 0, 0, Ss, Hh);
  wmma_gemm_kernel<float, float><<<gblocks(Mrows, Rr), 128, 0, stream>>>(
      x, wqdT, cqraw, Mrows, Rr, Dd, 0, 0, 0, Ss, Hh);
  // RMS norms -> bf16 latents
  rmsnorm_kernel<<<(Mrows * 32 + 255) / 256, 256, 0, stream>>>(ckvraw, kv_norm_w, ckv, Mrows);
  rmsnorm_kernel<<<(Mrows * 32 + 255) / 256, 256, 0, stream>>>(cqraw, q_norm_w, cq, Mrows);
  // Up projections scattered into attention layouts
  wmma_gemm_kernel<bf16, bf16><<<gblocks(Mrows, 512), 128, 0, stream>>>(
      ckv, wukT, Kb, Mrows, 512, Rr, 1, 32, 0, Ss, Hh);   // k_nope -> K[...,0:32]
  wmma_gemm_kernel<bf16, bf16><<<gblocks(Mrows, 512), 128, 0, stream>>>(
      ckv, wurT, Kb, Mrows, 512, Rr, 1, 32, 32, Ss, Hh);  // k_rope -> K[...,32:64]
  wmma_gemm_kernel<bf16, bf16><<<gblocks(Mrows, 1024), 128, 0, stream>>>(
      ckv, wuvT, VTb, Mrows, 1024, Rr, 2, 64, 0, Ss, Hh); // v -> VT (B,H,64,S)
  wmma_gemm_kernel<bf16, bf16><<<gblocks(Mrows, 512), 128, 0, stream>>>(
      cq, wuqT, Qb, Mrows, 512, Rr, 1, 32, 0, Ss, Hh);    // q_nope
  wmma_gemm_kernel<bf16, bf16><<<gblocks(Mrows, 512), 128, 0, stream>>>(
      cq, wqrT, Qb, Mrows, 512, Rr, 1, 32, 32, Ss, Hh);   // q_rope
  // RoPE on rope halves of Q and K
  rope_kernel<<<(Bb * Hh * Ss * 16 + 255) / 256, 256, 0, stream>>>(Qb, Kb, cosb, sinb, Ss);
  // Flash attention (transposed, LDS-free, split masked/unmasked loops)
  dim3 agrid(Ss / 64, Bb * Hh);
  mla_attention_kernel<<<agrid, 128, 0, stream>>>(Qb, Kb, VTb, attout, Ss, Hh);
  // Output projection -> f32
  wmma_gemm_kernel<bf16, float><<<gblocks(Mrows, Dd), 128, 0, stream>>>(
      attout, woT, out, Mrows, Dd, 1024, 0, 0, 0, Ss, Hh);
}
